// MultiLoRALinear_73967926772410
// MI455X (gfx1250) — matrix operand order repaired and checked by reading, not compiled, not run on hardware
//
#include <hip/hip_runtime.h>
#include <hip/hip_bf16.h>
#include <stdint.h>

// ---------------------------------------------------------------------------
// MultiLoRALinear on gfx1250 (MI455X):
//   out = X @ W^T + bias + (X @ A[idx]) @ B[idx]
// Base GEMM (M=16384,N=4096,K=4096, f32) via split-precision 3xBF16 on
// v_wmma_f32_16x16x32_bf16 (f32 accumulate).  K-loop pipelined with
// GLOBAL_LOAD_ASYNC_TO_LDS_B128 double buffering (ASYNCcnt).
// LoRA (rank 16) fused in the epilogue from inter = X @ A[idx] (d_ws).
// ---------------------------------------------------------------------------

typedef __attribute__((ext_vector_type(16))) __bf16 v16bf;
typedef __attribute__((ext_vector_type(8)))  __bf16 v8bf;
typedef __attribute__((ext_vector_type(8)))  float  v8f;
typedef __attribute__((ext_vector_type(4)))  float  f32x4;
typedef __attribute__((ext_vector_type(4)))  int    v4i;

#define IN_F   4096
#define OUT_F  4096
#define RANK   16
#define SEQ    2048
#define MTOT   16384   // 8 * 2048

#define CAT16(lo, hi) __builtin_shufflevector(lo, hi, 0,1,2,3,4,5,6,7,8,9,10,11,12,13,14,15)

#if __has_builtin(__builtin_amdgcn_global_load_async_to_lds_b128) && \
    __has_builtin(__builtin_amdgcn_s_wait_asynccnt)
#define USE_ASYNC 1
#else
#define USE_ASYNC 0
#endif

#if USE_ASYNC
// builtin signature (from clang diagnostic): param0 = v4i addrspace(1)*,
// param1 = v4i addrspace(3)*, then two immediates (offset, cpol).
typedef __attribute__((address_space(1))) v4i* as1_v4i;
typedef __attribute__((address_space(3))) v4i* as3_v4i;
#define AS1C(p) ((as1_v4i)(uintptr_t)(p))
#define AS3C(p) ((as3_v4i)(uint32_t)(uintptr_t)(p))
#endif

// Dynamic LDS layout for the GEMM kernel (bytes):
//   Xraw  [2][128][32] f32   @      0  (32768)   async double buffer
//   Wraw  [2][128][32] f32   @  32768  (32768)
//   Xh    [128][40]   bf16   @  65536  (10240)   row stride 80B, 16B aligned
//   Xl    [128][40]   bf16   @  75776  (10240)
//   Wh    [128][40]   bf16   @  86016  (10240)
//   Wl    [128][40]   bf16   @  96256  (10240)
//   interS[128][16]   f32    @ 106496  ( 8192)
//   Bs    [16][132]   f32    @ 114688  ( 8448)   row stride 528B (33x16)
#define SMEM_BYTES 123136

// ---------------------------------------------------------------------------
// Kernel 1: inter[m, r] = sum_k x[m, k] * A[idx[b(m)]][k][r]
// Memory-bound at rank 16 -> VALU with X tiled in LDS.
// ---------------------------------------------------------------------------
__global__ __launch_bounds__(256) void lora_inter_kernel(
    const float* __restrict__ x, const float* __restrict__ loraA,
    const int* __restrict__ idx, float* __restrict__ inter)
{
  __shared__ float xt[32][68];

  const int tid  = threadIdx.x;
  const int m0   = blockIdx.x * 32;
  const int slot = idx[m0 / SEQ];
  const float* __restrict__ A = loraA + (size_t)slot * IN_F * RANK;

  const int r    = tid & 15;
  const int rp   = tid >> 4;
  const int row0 = rp * 2, row1 = rp * 2 + 1;
  float acc0 = 0.f, acc1 = 0.f;

  const int lrow = tid >> 3;
  const int g    = (tid & 7) * 8;

  for (int kc = 0; kc < IN_F; kc += 64) {
    __syncthreads();
    const float* src = x + (size_t)(m0 + lrow) * IN_F + kc + g;
    *(f32x4*)&xt[lrow][g]     = *(const f32x4*)(src);
    *(f32x4*)&xt[lrow][g + 4] = *(const f32x4*)(src + 4);
    __syncthreads();

    const float* Ak = A + (size_t)kc * RANK + r;
#pragma unroll 8
    for (int kk = 0; kk < 64; ++kk) {
      const float aval = Ak[kk * RANK];
      acc0 = fmaf(xt[row0][kk], aval, acc0);
      acc1 = fmaf(xt[row1][kk], aval, acc1);
    }
  }
  inter[(size_t)(m0 + row0) * RANK + r] = acc0;
  inter[(size_t)(m0 + row1) * RANK + r] = acc1;
}

// ---------------------------------------------------------------------------
// Kernel 2: out = X @ W^T + bias + inter @ B[idx]
// Grid (N/128, M/128) = (32, 128); 256 threads = 8 waves.
// Wave w owns rows [w*16, w*16+16) of the 128x128 tile: 8 v8f accumulators.
// ---------------------------------------------------------------------------
__global__ __launch_bounds__(256) void multilora_gemm_kernel(
    const float* __restrict__ x, const float* __restrict__ w,
    const float* __restrict__ bias, const float* __restrict__ loraB,
    const int* __restrict__ idx, const float* __restrict__ inter,
    float* __restrict__ out)
{
  extern __shared__ char smem[];
  float*  Xraw   = (float*) (smem);            // [2][128][32]
  float*  Wraw   = (float*) (smem + 32768);    // [2][128][32]
  __bf16* Xh     = (__bf16*)(smem + 65536);    // [128][40]
  __bf16* Xl     = (__bf16*)(smem + 75776);
  __bf16* Wh     = (__bf16*)(smem + 86016);
  __bf16* Wl     = (__bf16*)(smem + 96256);
  float*  interS = (float*) (smem + 106496);   // [128][16]
  float*  Bs     = (float*) (smem + 114688);   // [16][132]

  const int tid  = threadIdx.x;
  const int n0   = blockIdx.x * 128;
  const int m0   = blockIdx.y * 128;           // 128 | 2048 -> one batch/block
  const int slot = idx[m0 / SEQ];

  // ---- stage LoRA operands for the epilogue -------------------------------
  {
    const int ir = tid >> 1, seg = (tid & 1) * 8;
    *(f32x4*)(interS + ir * RANK + seg)     =
        *(const f32x4*)(inter + (size_t)(m0 + ir) * RANK + seg);
    *(f32x4*)(interS + ir * RANK + seg + 4) =
        *(const f32x4*)(inter + (size_t)(m0 + ir) * RANK + seg + 4);
    const int br = tid >> 4, cg8 = (tid & 15) * 8;
    const float* Bsrc = loraB + (size_t)slot * RANK * OUT_F + (size_t)br * OUT_F + n0 + cg8;
    *(f32x4*)(Bs + br * 132 + cg8)     = *(const f32x4*)(Bsrc);
    *(f32x4*)(Bs + br * 132 + cg8 + 4) = *(const f32x4*)(Bsrc + 4);
  }

  const int wave = tid >> 5;
  const int lane = tid & 31;
  const int lm   = lane & 15;
  const int lh   = lane >> 4;

  v8f c[8];
#pragma unroll
  for (int i = 0; i < 8; ++i)
#pragma unroll
    for (int j = 0; j < 8; ++j) c[i][j] = 0.f;

  const int lrow = tid >> 1;            // converter mapping: 16 elems/thread
  const int cg   = (tid & 1) * 16;

#if USE_ASYNC
  // ---- async prologue: tile kc=0 into buffer 0 (8 b128 per thread) --------
#pragma unroll
  for (int l = 0; l < 4; ++l) {
    const int e   = l * 256 + tid;      // 1024 16-byte chunks per operand
    const int row = e >> 3;
    const int c4  = (e & 7) * 4;
    __builtin_amdgcn_global_load_async_to_lds_b128(
        AS1C(x + (size_t)(m0 + row) * IN_F + c4), AS3C(Xraw + row * 32 + c4), 0, 0);
    __builtin_amdgcn_global_load_async_to_lds_b128(
        AS1C(w + (size_t)(n0 + row) * IN_F + c4), AS3C(Wraw + row * 32 + c4), 0, 0);
  }
  int p = 0;
#else
  const float* xrow = x + (size_t)(m0 + lrow) * IN_F + cg;
  const float* wrow = w + (size_t)(n0 + lrow) * IN_F + cg;
#endif

  for (int kc = 0; kc < IN_F; kc += 32) {
#if USE_ASYNC
    // ---- issue next tile, then drain current (8 next-tile ops in flight) --
    if (kc + 32 < IN_F) {
      const int nb = p ^ 1;
#pragma unroll
      for (int l = 0; l < 4; ++l) {
        const int e   = l * 256 + tid;
        const int row = e >> 3;
        const int c4  = (e & 7) * 4;
        __builtin_amdgcn_global_load_async_to_lds_b128(
            AS1C(x + (size_t)(m0 + row) * IN_F + (kc + 32) + c4),
            AS3C(Xraw + nb * 4096 + row * 32 + c4), 0, 0);
        __builtin_amdgcn_global_load_async_to_lds_b128(
            AS1C(w + (size_t)(n0 + row) * IN_F + (kc + 32) + c4),
            AS3C(Wraw + nb * 4096 + row * 32 + c4), 0, 0);
      }
      __builtin_amdgcn_s_wait_asynccnt(8);
    } else {
      __builtin_amdgcn_s_wait_asynccnt(0);
    }
    __syncthreads();
    // ---- convert pass: raw f32 (LDS) -> bf16 hi/lo (LDS), once per elem ---
    {
      const float* xsrc = Xraw + p * 4096 + lrow * 32 + cg;
      const float* wsrc = Wraw + p * 4096 + lrow * 32 + cg;
#pragma unroll
      for (int j = 0; j < 16; j += 4) {
        const f32x4 xv = *(const f32x4*)(xsrc + j);
        const f32x4 wv = *(const f32x4*)(wsrc + j);
#pragma unroll
        for (int e = 0; e < 4; ++e) {
          const float  xf  = xv[e];
          const __bf16 xhi = (__bf16)xf;
          Xh[lrow * 40 + cg + j + e] = xhi;
          Xl[lrow * 40 + cg + j + e] = (__bf16)(xf - (float)xhi);
          const float  wf  = wv[e];
          const __bf16 whi = (__bf16)wf;
          Wh[lrow * 40 + cg + j + e] = whi;
          Wl[lrow * 40 + cg + j + e] = (__bf16)(wf - (float)whi);
        }
      }
    }
    __syncthreads();
#else
    // ---- sync fallback: global f32 -> convert -> bf16 LDS -----------------
    __syncthreads();
#pragma unroll
    for (int j = 0; j < 16; j += 4) {
      const f32x4 xv = *(const f32x4*)(xrow + kc + j);
      const f32x4 wv = *(const f32x4*)(wrow + kc + j);
#pragma unroll
      for (int e = 0; e < 4; ++e) {
        const float  xf  = xv[e];
        const __bf16 xhi = (__bf16)xf;
        Xh[lrow * 40 + cg + j + e] = xhi;
        Xl[lrow * 40 + cg + j + e] = (__bf16)(xf - (float)xhi);
        const float  wf  = wv[e];
        const __bf16 whi = (__bf16)wf;
        Wh[lrow * 40 + cg + j + e] = whi;
        Wl[lrow * 40 + cg + j + e] = (__bf16)(wf - (float)whi);
      }
    }
    __syncthreads();
#endif

    // ---- WMMA phase -------------------------------------------------------
    // A frag (16x32 bf16): lane = row M; lanes 0-15 K {0-7,16-23},
    // lanes 16-31 K {8-15,24-31}.
    const int arow = wave * 16 + lm;
    const int koff = lh * 8;
    const v16bf a_hi = CAT16(*(const v8bf*)(Xh + arow * 40 + koff),
                             *(const v8bf*)(Xh + arow * 40 + koff + 16));
    const v16bf a_lo = CAT16(*(const v8bf*)(Xl + arow * 40 + koff),
                             *(const v8bf*)(Xl + arow * 40 + koff + 16));

#pragma unroll
    for (int ns = 0; ns < 8; ++ns) {
      // B frag (32x16 bf16): lane = col N (= W row); lanes 0-15 K {0-7,16-23},
      // lanes 16-31 K {8-15,24-31}.
      const int brow = ns * 16 + lm;
      const int kb   = lh * 8;
      const v16bf b_hi = CAT16(*(const v8bf*)(Wh + brow * 40 + kb),
                               *(const v8bf*)(Wh + brow * 40 + kb + 16));
      const v16bf b_lo = CAT16(*(const v8bf*)(Wl + brow * 40 + kb),
                               *(const v8bf*)(Wl + brow * 40 + kb + 16));
      c[ns] = __builtin_amdgcn_wmma_f32_16x16x32_bf16(false, a_hi, false, b_hi,
                                                      (short)0, c[ns], false, false);
      c[ns] = __builtin_amdgcn_wmma_f32_16x16x32_bf16(false, a_hi, false, b_lo,
                                                      (short)0, c[ns], false, false);
      c[ns] = __builtin_amdgcn_wmma_f32_16x16x32_bf16(false, a_lo, false, b_hi,
                                                      (short)0, c[ns], false, false);
    }
#if USE_ASYNC
    p ^= 1;
#endif
  }

  // ---- epilogue: + bias + rank-16 LoRA, single store ----------------------
  // C/D layout: VGPR v of lane L holds (M = v + 8*(L>=16), N = L%16).
  float bv[8];
#pragma unroll
  for (int ns = 0; ns < 8; ++ns) bv[ns] = bias[n0 + ns * 16 + lm];

#pragma unroll
  for (int v = 0; v < 8; ++v) {
    const int mloc = wave * 16 + v + 8 * lh;
    float ir[RANK];
#pragma unroll
    for (int r = 0; r < RANK; ++r) ir[r] = interS[mloc * RANK + r];
    float* orow = out + (size_t)(m0 + mloc) * OUT_F + n0 + lm;
#pragma unroll
    for (int ns = 0; ns < 8; ++ns) {
      const int nloc = ns * 16 + lm;
      float lora = 0.f;
#pragma unroll
      for (int r = 0; r < RANK; ++r) lora = fmaf(ir[r], Bs[r * 132 + nloc], lora);
      orow[ns * 16] = c[ns][v] + bv[ns] + lora;   // ALPHA = 1.0
    }
  }
}

// ---------------------------------------------------------------------------
extern "C" void kernel_launch(void* const* d_in, const int* in_sizes, int n_in,
                              void* d_out, int out_size, void* d_ws, size_t ws_size,
                              hipStream_t stream) {
  (void)in_sizes; (void)n_in; (void)out_size; (void)ws_size;
  const float* x     = (const float*)d_in[0];
  const float* wgt   = (const float*)d_in[1];
  const float* bias  = (const float*)d_in[2];
  const float* loraA = (const float*)d_in[3];
  const float* loraB = (const float*)d_in[4];
  const int*   idx   = (const int*)d_in[5];
  float* out   = (float*)d_out;
  float* inter = (float*)d_ws;                    // 16384*16*4 = 1 MiB scratch

  lora_inter_kernel<<<MTOT / 32, 256, 0, stream>>>(x, loraA, idx, inter);
  multilora_gemm_kernel<<<dim3(OUT_F / 128, MTOT / 128), 256, SMEM_BYTES, stream>>>(
      x, wgt, bias, loraB, idx, inter, out);
}
